// GPT_15762529976868
// MI455X (gfx1250) — compile-verified, matrix operand order checked by reading, not run on hardware
//
#include <hip/hip_runtime.h>
#include <hip/hip_bf16.h>
#include <cstdint>

#define DEV __device__ __forceinline__

typedef __attribute__((ext_vector_type(16))) _Float16 v16h;
typedef __attribute__((ext_vector_type(8)))  float    v8f;

union F16Frag { v16h v; unsigned u[8]; _Float16 h[16]; };
union F32Acc  { v8f v; float f[8]; };

constexpr int Bc = 256, Tc = 256, Vc = 128, Dc = 384, Hc = 6, Lc = 6, HSc = 64, FFc = 1536;
constexpr int BT = Bc * Tc;

// k-offset inside a 16x32 f16 A/B fragment (CDNA5 ISA 7.12.2):
// lanes 0-15 -> khalf=0 (K 0..7 then 16..23), lanes 16-31 -> khalf=1 (K 8..15 then 24..31)
DEV int kOff(int j, int khalf) { return ((j & 4) << 2) + (khalf << 3) + ((j & 3) << 1); }

DEV v16h loadFrag(const _Float16* row, int khalf) {
  F16Frag f;
#pragma unroll
  for (int j = 0; j < 8; ++j)
    f.u[j] = *reinterpret_cast<const unsigned*>(row + kOff(j, khalf));
  return f.v;
}

DEV v8f wmma(v16h a, v16h b, v8f c) {
  return __builtin_amdgcn_wmma_f32_16x16x32_f16(false, a, false, b, (short)0, c, false, false);
}

// CDNA5 async global->LDS copy (ASYNCcnt-tracked). 16B per lane.
// vdst = per-lane LDS byte address (generic-addr low 32 bits), vaddr = 64-bit global addr.
DEV void asyncLoadB128(const _Float16* g, _Float16* l) {
  unsigned      lo = (unsigned)(unsigned long)l;          // LDS offset (aperture low bits)
  unsigned long ga = (unsigned long)g;
  asm volatile("global_load_async_to_lds_b128 %0, %1, off"
               :: "v"(lo), "v"(ga) : "memory");
}
DEV void waitAsync3() { asm volatile("s_wait_asynccnt 3" ::: "memory"); }
DEV void waitAsync0() { asm volatile("s_wait_asynccnt 0" ::: "memory"); }

// ---------------- embedding ----------------
__global__ __launch_bounds__(256) void embed_kernel(const int* __restrict__ idx,
                                                    const float* __restrict__ tok,
                                                    const float* __restrict__ pos,
                                                    float* __restrict__ x) {
  size_t gid = (size_t)blockIdx.x * blockDim.x + threadIdx.x;
  if (gid >= (size_t)BT * Dc) return;
  int d = (int)(gid % Dc);
  size_t bt = gid / Dc;
  int t = (int)(bt % Tc);
  int tk = idx[bt];
  x[gid] = tok[(size_t)tk * Dc + d] + pos[(size_t)t * Dc + d];
}

// ---------------- weight converts (fp32 -> f16, transposed to [N][K]) ----------------
__global__ __launch_bounds__(256) void cvt_qkv_kernel(const float* __restrict__ Wq,
                                                      const float* __restrict__ Wk,
                                                      const float* __restrict__ Wv,
                                                      _Float16* __restrict__ dst) {
  size_t gid = (size_t)blockIdx.x * blockDim.x + threadIdx.x;
  size_t total = (size_t)Lc * 3 * Hc * Dc * HSc;
  if (gid >= total) return;
  int d = (int)(gid % Dc); size_t r = gid / Dc;
  int hs = (int)(r % HSc); r /= HSc;
  int h  = (int)(r % Hc);  r /= Hc;
  int which = (int)(r % 3);
  int l = (int)(r / 3);
  const float* src = (which == 0) ? Wq : (which == 1) ? Wk : Wv;
  float val = src[(((size_t)l * Hc + h) * Dc + d) * HSc + hs];
  dst[((size_t)l * 3 * Dc + (size_t)which * Dc + h * HSc + hs) * Dc + d] = (_Float16)val;
}

__global__ __launch_bounds__(256) void cvt_T_kernel(const float* __restrict__ src,
                                                    _Float16* __restrict__ dst,
                                                    int K, int N, long total) {
  long gid = (long)blockIdx.x * blockDim.x + threadIdx.x;
  if (gid >= total) return;
  long per = (long)K * N;
  long e = gid % per;
  long bidx = gid / per;
  int n = (int)(e / K);
  int k = (int)(e % K);
  dst[bidx * per + (size_t)n * K + k] = (_Float16)src[bidx * per + (size_t)k * N + n];
}

// ---------------- layernorm (one row per 128-thread block) ----------------
__global__ __launch_bounds__(128) void ln_kernel(const float* __restrict__ x,
                                                 const float* __restrict__ g,
                                                 const float* __restrict__ bta,
                                                 _Float16* __restrict__ out) {
  int row = blockIdx.x;
  int t = threadIdx.x;
  const float* xr = x + (size_t)row * Dc;
  float v0 = xr[t], v1 = xr[t + 128], v2 = xr[t + 256];
  float s = v0 + v1 + v2;
  float s2 = v0 * v0 + v1 * v1 + v2 * v2;
#pragma unroll
  for (int off = 16; off; off >>= 1) {
    s  += __shfl_xor(s,  off, 32);
    s2 += __shfl_xor(s2, off, 32);
  }
  __shared__ float sh[8];
  int wave = t >> 5;
  if ((t & 31) == 0) { sh[wave] = s; sh[4 + wave] = s2; }
  __syncthreads();
  s  = sh[0] + sh[1] + sh[2] + sh[3];
  s2 = sh[4] + sh[5] + sh[6] + sh[7];
  float mean = s * (1.f / Dc);
  float var  = s2 * (1.f / Dc) - mean * mean;
  float rstd = rsqrtf(var + 1e-5f);
  _Float16* o = out + (size_t)row * Dc;
  o[t]       = (_Float16)((v0 - mean) * rstd * g[t]       + bta[t]);
  o[t + 128] = (_Float16)((v1 - mean) * rstd * g[t + 128] + bta[t + 128]);
  o[t + 256] = (_Float16)((v2 - mean) * rstd * g[t + 256] + bta[t + 256]);
}

// ---------------- LDS-staged, async-double-buffered WMMA GEMM ----------------
// Block = 8 waves computes a 128(M) x 64(N) tile; K stepped by 32.
// A[M,K] f16 row-major; Bt[N,K] f16 (pre-transposed weights).
// MODE 0: out f16 + ReLU   MODE 1: fp32 residual add   MODE 2: fp32 + bias
// MODE 3: fused QKV scatter (q,k -> [B,H,T,HS]; v -> [B,H,HS,T])
template <int MODE>
__global__ __launch_bounds__(256) void wmma_gemm_lds(const _Float16* __restrict__ A,
                                                     const _Float16* __restrict__ Bt,
                                                     const float* __restrict__ bias,
                                                     float* __restrict__ outF,
                                                     _Float16* __restrict__ outH,
                                                     _Float16* __restrict__ qo,
                                                     _Float16* __restrict__ ko,
                                                     _Float16* __restrict__ vo,
                                                     int N, int K) {
  __shared__ _Float16 As[2][128 * 32];   // 2 x 8KB
  __shared__ _Float16 Bs[2][64 * 32];    // 2 x 4KB
  const int tid = threadIdx.x;
  const int lane = tid & 31;
  const int wave = tid >> 5;
  const int l16 = lane & 15;
  const int khalf = lane >> 4;
  const int nBase = blockIdx.x * 64;
  const int mBase = blockIdx.y * 128;

  // per-thread 16B staging chunks: A = 512 chunks (2/thread), B = 256 chunks (1/thread)
  const int ar0 = tid >> 2;              // rows 0..63
  const int ar1 = ar0 + 64;              // rows 64..127
  const int asub = (tid & 3) * 8;        // f16 element offset within the 32-wide k-slab
  const _Float16* Ag0 = A + (size_t)(mBase + ar0) * K + asub;
  const _Float16* Ag1 = A + (size_t)(mBase + ar1) * K + asub;
  const _Float16* Bg  = Bt + (size_t)(nBase + ar0) * K + asub;

  const int nStages = K / 32;
  // prologue: stage 0 -> buffer 0
  asyncLoadB128(Ag0, &As[0][ar0 * 32 + asub]);
  asyncLoadB128(Ag1, &As[0][ar1 * 32 + asub]);
  asyncLoadB128(Bg,  &Bs[0][ar0 * 32 + asub]);

  F32Acc acc[4];
#pragma unroll
  for (int c = 0; c < 4; ++c)
#pragma unroll
    for (int i = 0; i < 8; ++i) acc[c].f[i] = 0.f;

  for (int s = 0; s < nStages; ++s) {
    const int buf = s & 1;
    if (s + 1 < nStages) {
      const int k0 = (s + 1) * 32;
      asyncLoadB128(Ag0 + k0, &As[buf ^ 1][ar0 * 32 + asub]);
      asyncLoadB128(Ag1 + k0, &As[buf ^ 1][ar1 * 32 + asub]);
      asyncLoadB128(Bg + k0,  &Bs[buf ^ 1][ar0 * 32 + asub]);
      waitAsync3();   // async ops complete in order: stage s's 3 are done
    } else {
      waitAsync0();
    }
    __syncthreads();
    v16h a = loadFrag(&As[buf][(wave * 16 + l16) * 32], khalf);
#pragma unroll
    for (int c = 0; c < 4; ++c) {
      v16h b = loadFrag(&Bs[buf][(c * 16 + l16) * 32], khalf);
      acc[c].v = wmma(a, b, acc[c].v);
    }
    __syncthreads();   // all waves done reading buf before it is refilled
  }

  // epilogue
#pragma unroll
  for (int c = 0; c < 4; ++c) {
    const int n = nBase + c * 16 + l16;
    float bv = 0.f;
    if (MODE != 3) bv = bias ? bias[n] : 0.f;
    int which = 0, h = 0, hs = 0;
    if (MODE == 3) { which = n / Dc; int rem = n % Dc; h = rem >> 6; hs = rem & 63; }
#pragma unroll
    for (int r = 0; r < 8; ++r) {
      const size_t row = (size_t)mBase + wave * 16 + khalf * 8 + r;
      float val = acc[c].f[r] + bv;
      if (MODE == 0)      outH[row * (size_t)N + n] = (_Float16)fmaxf(val, 0.f);
      else if (MODE == 1) outF[row * (size_t)N + n] += val;
      else if (MODE == 2) outF[row * (size_t)N + n] = val;
      else {
        const int b = (int)(row / Tc);
        const int t = (int)(row % Tc);
        const _Float16 hv = (_Float16)val;
        if (which == 0)      qo[(((size_t)b * Hc + h) * Tc + t) * HSc + hs] = hv;
        else if (which == 1) ko[(((size_t)b * Hc + h) * Tc + t) * HSc + hs] = hv;
        else                 vo[(((size_t)b * Hc + h) * HSc + hs) * Tc + t] = hv;
      }
    }
  }
}

// ---------------- flash attention: one wave32 per (b, h, 16-query tile) ----------------
__global__ __launch_bounds__(32) void attn_kernel(const _Float16* __restrict__ q,
                                                  const _Float16* __restrict__ kmat,
                                                  const _Float16* __restrict__ vT,
                                                  _Float16* __restrict__ out) {
  int qt = blockIdx.x;   // T/16 tiles
  int h  = blockIdx.y;
  int b  = blockIdx.z;
  int lane = threadIdx.x & 31;
  int l16 = lane & 15;
  int khalf = lane >> 4;
  const float scale = 0.125f;   // HS^-0.5 = 1/8
  __shared__ _Float16 pbuf[16 * 32];

  const size_t bh = (size_t)b * Hc + h;
  const _Float16* qrow = q + (bh * Tc + (size_t)qt * 16 + l16) * HSc;
  v16h qf0 = loadFrag(qrow, khalf);
  v16h qf1 = loadFrag(qrow + 32, khalf);

  float mrow[8], lrow[8];
  F32Acc O[4];
#pragma unroll
  for (int r = 0; r < 8; ++r) { mrow[r] = -3.0e38f; lrow[r] = 0.f; }
#pragma unroll
  for (int c = 0; c < 4; ++c)
#pragma unroll
    for (int r = 0; r < 8; ++r) O[c].f[r] = 0.f;

  int maxkey = qt * 16 + 15;
  int nkb = maxkey / 32 + 1;
  for (int kb = 0; kb < nkb; ++kb) {
    F32Acc s0, s1;
#pragma unroll
    for (int i = 0; i < 8; ++i) { s0.f[i] = 0.f; s1.f[i] = 0.f; }
    const _Float16* kbase = kmat + (bh * Tc + (size_t)kb * 32) * HSc;
    const _Float16* kr0 = kbase + (size_t)l16 * HSc;
    const _Float16* kr1 = kbase + (size_t)(l16 + 16) * HSc;
    s0.v = wmma(qf0, loadFrag(kr0, khalf), s0.v);
    s1.v = wmma(qf0, loadFrag(kr1, khalf), s1.v);
    s0.v = wmma(qf1, loadFrag(kr0 + 32, khalf), s0.v);
    s1.v = wmma(qf1, loadFrag(kr1 + 32, khalf), s1.v);

    float alpha[8];
#pragma unroll
    for (int r = 0; r < 8; ++r) {
      int tq = qt * 16 + khalf * 8 + r;
      int key0 = kb * 32 + l16;
      int key1 = key0 + 16;
      bool vld0 = key0 <= tq, vld1 = key1 <= tq;
      float a0 = vld0 ? s0.f[r] * scale : -3.0e38f;
      float a1 = vld1 ? s1.f[r] * scale : -3.0e38f;
      float mx = fmaxf(a0, a1);
#pragma unroll
      for (int off = 8; off; off >>= 1) mx = fmaxf(mx, __shfl_xor(mx, off, 16));
      float newm = fmaxf(mrow[r], mx);
      bool any = newm > -1.0e37f;
      float al = any ? __expf(mrow[r] - newm) : 0.f;
      float p0 = (vld0 && any) ? __expf(a0 - newm) : 0.f;
      float p1 = (vld1 && any) ? __expf(a1 - newm) : 0.f;
      float ps = p0 + p1;
#pragma unroll
      for (int off = 8; off; off >>= 1) ps += __shfl_xor(ps, off, 16);
      lrow[r] = lrow[r] * al + ps;
      mrow[r] = newm;
      alpha[r] = al;
      int prow = khalf * 8 + r;
      pbuf[prow * 32 + l16]      = (_Float16)p0;
      pbuf[prow * 32 + l16 + 16] = (_Float16)p1;
    }
#pragma unroll
    for (int c = 0; c < 4; ++c)
#pragma unroll
      for (int r = 0; r < 8; ++r) O[c].f[r] *= alpha[r];
    __syncthreads();
    v16h pa = loadFrag(pbuf + (size_t)l16 * 32, khalf);
#pragma unroll
    for (int c = 0; c < 4; ++c) {
      const _Float16* vr = vT + (bh * HSc + (size_t)(c * 16 + l16)) * Tc + (size_t)kb * 32;
      O[c].v = wmma(pa, loadFrag(vr, khalf), O[c].v);
    }
    __syncthreads();
  }
  float inv[8];
#pragma unroll
  for (int r = 0; r < 8; ++r) inv[r] = lrow[r] > 0.f ? 1.f / lrow[r] : 0.f;
#pragma unroll
  for (int c = 0; c < 4; ++c)
#pragma unroll
    for (int r = 0; r < 8; ++r) {
      int tq = qt * 16 + khalf * 8 + r;
      int hs = c * 16 + l16;
      out[(((size_t)b * Tc + tq) * Hc + h) * HSc + hs] = (_Float16)(O[c].f[r] * inv[r]);
    }
}

// ---------------- host orchestration ----------------
extern "C" void kernel_launch(void* const* d_in, const int* in_sizes, int n_in,
                              void* d_out, int out_size, void* d_ws, size_t ws_size,
                              hipStream_t stream) {
  (void)in_sizes; (void)n_in; (void)out_size; (void)ws_size;
  const int*   idx   = (const int*)d_in[0];
  const float* tok   = (const float*)d_in[1];
  const float* pos   = (const float*)d_in[2];
  const float* Wq    = (const float*)d_in[3];
  const float* Wk    = (const float*)d_in[4];
  const float* Wv    = (const float*)d_in[5];
  const float* Wproj = (const float*)d_in[6];
  const float* bproj = (const float*)d_in[7];
  const float* ln1g  = (const float*)d_in[8];
  const float* ln1b  = (const float*)d_in[9];
  const float* ln2g  = (const float*)d_in[10];
  const float* ln2b  = (const float*)d_in[11];
  const float* W1    = (const float*)d_in[12];
  const float* b1    = (const float*)d_in[13];
  const float* W2    = (const float*)d_in[14];
  const float* b2    = (const float*)d_in[15];
  const float* lnfg  = (const float*)d_in[16];
  const float* lnfb  = (const float*)d_in[17];
  const float* Wlm   = (const float*)d_in[18];
  const float* blm   = (const float*)d_in[19];

  char* ws = (char*)d_ws;
  size_t off = 0;
  auto alloc = [&](size_t bytes) -> void* {
    void* p = ws + off;
    off += (bytes + 255) & ~(size_t)255;
    return p;
  };
  float*     x     = (float*)alloc((size_t)BT * Dc * 4);
  _Float16*  xn    = (_Float16*)alloc((size_t)BT * Dc * 2);
  _Float16*  qb    = (_Float16*)alloc((size_t)Bc * Hc * Tc * HSc * 2);
  _Float16*  kb    = (_Float16*)alloc((size_t)Bc * Hc * Tc * HSc * 2);
  _Float16*  vT    = (_Float16*)alloc((size_t)Bc * Hc * HSc * Tc * 2);
  _Float16*  attn  = (_Float16*)alloc((size_t)BT * Dc * 2);
  _Float16*  hbuf  = (_Float16*)alloc((size_t)BT * FFc * 2);
  _Float16*  WqkvT = (_Float16*)alloc((size_t)Lc * 3 * Dc * Dc * 2);
  _Float16*  WprojT= (_Float16*)alloc((size_t)Lc * Dc * Dc * 2);
  _Float16*  W1T   = (_Float16*)alloc((size_t)Lc * FFc * Dc * 2);
  _Float16*  W2T   = (_Float16*)alloc((size_t)Lc * Dc * FFc * 2);
  _Float16*  WlmT  = (_Float16*)alloc((size_t)Vc * Dc * 2);

  {
    long tot = (long)Lc * 3 * Hc * Dc * HSc;
    cvt_qkv_kernel<<<(unsigned)((tot + 255) / 256), 256, 0, stream>>>(Wq, Wk, Wv, WqkvT);
  }
  {
    long tot = (long)Lc * Dc * Dc;
    cvt_T_kernel<<<(unsigned)((tot + 255) / 256), 256, 0, stream>>>(Wproj, WprojT, Dc, Dc, tot);
  }
  {
    long tot = (long)Lc * Dc * FFc;
    cvt_T_kernel<<<(unsigned)((tot + 255) / 256), 256, 0, stream>>>(W1, W1T, Dc, FFc, tot);
    cvt_T_kernel<<<(unsigned)((tot + 255) / 256), 256, 0, stream>>>(W2, W2T, FFc, Dc, tot);
  }
  {
    long tot = (long)Dc * Vc;
    cvt_T_kernel<<<(unsigned)((tot + 255) / 256), 256, 0, stream>>>(Wlm, WlmT, Dc, Vc, tot);
  }
  {
    long tot = (long)BT * Dc;
    embed_kernel<<<(unsigned)((tot + 255) / 256), 256, 0, stream>>>(idx, tok, pos, x);
  }

  for (int l = 0; l < Lc; ++l) {
    ln_kernel<<<BT, 128, 0, stream>>>(x, ln1g + (size_t)l * Dc, ln1b + (size_t)l * Dc, xn);

    dim3 gq(3 * Dc / 64, BT / 128);
    wmma_gemm_lds<3><<<gq, 256, 0, stream>>>(xn, WqkvT + (size_t)l * 3 * Dc * Dc,
                                             nullptr, nullptr, nullptr, qb, kb, vT, 3 * Dc, Dc);

    dim3 ga(Tc / 16, Hc, Bc);
    attn_kernel<<<ga, 32, 0, stream>>>(qb, kb, vT, attn);

    dim3 gp(Dc / 64, BT / 128);
    wmma_gemm_lds<1><<<gp, 256, 0, stream>>>(attn, WprojT + (size_t)l * Dc * Dc,
                                             bproj + (size_t)l * Dc, x, nullptr,
                                             nullptr, nullptr, nullptr, Dc, Dc);

    ln_kernel<<<BT, 128, 0, stream>>>(x, ln2g + (size_t)l * Dc, ln2b + (size_t)l * Dc, xn);

    dim3 g1(FFc / 64, BT / 128);
    wmma_gemm_lds<0><<<g1, 256, 0, stream>>>(xn, W1T + (size_t)l * FFc * Dc,
                                             b1 + (size_t)l * FFc, nullptr, hbuf,
                                             nullptr, nullptr, nullptr, FFc, Dc);

    dim3 g2(Dc / 64, BT / 128);
    wmma_gemm_lds<1><<<g2, 256, 0, stream>>>(hbuf, W2T + (size_t)l * Dc * FFc,
                                             b2 + (size_t)l * Dc, x, nullptr,
                                             nullptr, nullptr, nullptr, Dc, FFc);
  }

  ln_kernel<<<BT, 128, 0, stream>>>(x, lnfg, lnfb, xn);
  dim3 gl(Vc / 64, BT / 128);
  wmma_gemm_lds<2><<<gl, 256, 0, stream>>>(xn, WlmT, blm, (float*)d_out, nullptr,
                                           nullptr, nullptr, nullptr, Vc, Dc);
}
